// VelocityMLP_35304631173690
// MI455X (gfx1250) — compile-verified, matrix operand order checked by reading, not run on hardware
//
#include <hip/hip_runtime.h>
#include <math.h>

typedef __attribute__((ext_vector_type(16))) _Float16 v16h;
typedef __attribute__((ext_vector_type(8)))  float    v8f;

#define HID     128
#define WSTR    136   // weight LDS row stride (halves): 272B rows -> 16B-aligned b128, conflict-free
#define ASTR    136   // activation LDS row stride (halves): same properties
#define NWAVES  3
#define MPW     32    // points per wave: two 16-point WMMA N-tiles share each weight (A) fragment
#define BLOCK   (NWAVES * 32)
#define PTSPB   (NWAVES * MPW)

// gfx1250 v_tanh_f32 trans op (probe-confirmed).
__device__ __forceinline__ float fast_tanh(float v) {
#if __has_builtin(__builtin_amdgcn_tanhf)
  return __builtin_amdgcn_tanhf(v);
#else
  return 1.0f - __fdividef(2.0f, __expf(2.0f * v) + 1.0f);
#endif
}

__device__ __forceinline__ float fast_sigmoid(float z) {
  return __fdividef(1.0f, 1.0f + __expf(-z));
}

// pack two f32 into packed f16 (v_cvt_pk_rtz_f16_f32)
__device__ __forceinline__ unsigned int pk16(float a, float b) {
  auto t = __builtin_amdgcn_cvt_pkrtz(a, b);   // __fp16 ext_vector(2)
  return __builtin_bit_cast(unsigned int, t);
}

union FragU { v16h h; uint4 q[2]; };
union FragS { uint4 q; _Float16 hh[8]; };

// CDNA5 16-bit A/B fragment: lane<16 holds K in {0..7,16..23}, lane>=16 holds
// K in {8..15,24..31} of its row -> exactly two contiguous 16B runs per lane.
__device__ __forceinline__ void load_frag_lds(const _Float16* base, int row, int rstride,
                                              int kofs, int hi, FragU& f) {
  const _Float16* p = base + row * rstride + kofs + (hi << 3);
  f.q[0] = *reinterpret_cast<const uint4*>(p);        // K pairs 0..7   (this half)
  f.q[1] = *reinterpret_cast<const uint4*>(p + 16);   // K pairs 16..23 (this half)
}

// One-shot: convert + transpose the three 128x128 f32 hidden-layer weights
// into dense f16 W^T tiles in d_ws (m-th matrix at ws + m*16384, layout [n][k]).
__global__ void convert_weights_kernel(const float* __restrict__ W2,
                                       const float* __restrict__ W3,
                                       const float* __restrict__ W4,
                                       _Float16* __restrict__ ws)
{
  int idx = blockIdx.x * blockDim.x + threadIdx.x;
  if (idx >= 3 * HID * HID) return;
  int m = idx >> 14;
  int r = idx & (HID * HID - 1);
  int k = r >> 7, n = r & 127;
  const float* W = (m == 0) ? W2 : (m == 1) ? W3 : W4;
  ws[m * HID * HID + n * HID + k] = (_Float16)W[r];
}

__global__ __launch_bounds__(BLOCK)
void velocity_mlp_kernel(const float* __restrict__ xy, const float* __restrict__ U,
                         const float* __restrict__ W1, const float* __restrict__ b1,
                         const float* __restrict__ b2, const float* __restrict__ b3,
                         const float* __restrict__ b4,
                         const float* __restrict__ W5, const float* __restrict__ b5,
                         const _Float16* __restrict__ wts,   // pre-converted W^T f16, 3 matrices
                         float* __restrict__ out, int npts)
{
  extern __shared__ char smem[];
  _Float16* wt     = (_Float16*)smem;                                       // 128*WSTR halves
  _Float16* act    = (_Float16*)(smem + HID * WSTR * 2);                    // NWAVES*MPW*ASTR halves
  float*    sbias  = (float*)(smem + HID * WSTR * 2 + NWAVES * MPW * ASTR * 2); // b2,b3,b4

  const int tid  = threadIdx.x;
  const int wave = tid >> 5;
  const int lane = tid & 31;
  const int M    = lane & 15;
  const int hi   = lane >> 4;

  // ---- stage hidden-layer biases (vector-indexed) into LDS ----
  for (int i = tid; i < 128; i += BLOCK) {
    sbias[i]       = b2[i];
    sbias[128 + i] = b3[i];
    sbias[256 + i] = b4[i];
  }

  // ---- stage W2^T: pure b128 copies (pre-converted f16 in global scratch) ----
  {
    const uint4* src = (const uint4*)wts;
    for (int i = tid; i < HID * HID / 8; i += BLOCK) {
      int row = i >> 4, j = i & 15;
      *(uint4*)(wt + row * WSTR + j * 8) = src[i];
    }
  }
  __syncthreads();

  _Float16* actw = act + wave * (MPW * ASTR);
  int gp  = blockIdx.x * PTSPB + wave * MPW + lane;   // one point per lane
  int gpc = gp < npts ? gp : npts - 1;
  float x = xy[2 * gpc], y = xy[2 * gpc + 1];

  // ---- layer 1: h = tanh(xy @ W1 + b1); W1/b1 via uniform (scalar) loads,
  //      packed b128 stores of 8 f16 at a time ----
  for (int n0 = 0; n0 < HID; n0 += 8) {
    float t[8];
#pragma unroll
    for (int j = 0; j < 8; ++j) {
      float v = fmaf(x, W1[n0 + j], fmaf(y, W1[HID + n0 + j], b1[n0 + j]));
      t[j] = fast_tanh(v);
    }
    uint4 st;
    st.x = pk16(t[0], t[1]); st.y = pk16(t[2], t[3]);
    st.z = pk16(t[4], t[5]); st.w = pk16(t[6], t[7]);
    *reinterpret_cast<uint4*>(actw + lane * ASTR + n0) = st;
  }
  __syncthreads();

  // ---- layers 2..4 on WMMA: D^T = W^T x H^T  (A = weight tile, B = act tile).
  //      C layout then gives each lane 8 contiguous output neurons of one
  //      point -> single b128 store per tile. ----
  for (int L = 0; L < 3; ++L) {
    const float* bias = sbias + L * 128;

    if (L < 2) {  // warm L1 for the next restage while we compute
      const _Float16* Wn = wts + (L + 1) * HID * HID;
      for (int i = tid * 32; i < HID * HID; i += BLOCK * 32)
        __builtin_prefetch(Wn + i, 0, 0);
    }

    // B fragments: H^T for the two 16-point tiles, held in registers
    FragU bh0[4], bh1[4];
#pragma unroll
    for (int kt = 0; kt < 4; ++kt) {
      load_frag_lds(actw, M,      ASTR, kt * 32, hi, bh0[kt]);   // points 0..15
      load_frag_lds(actw, M + 16, ASTR, kt * 32, hi, bh1[kt]);   // points 16..31
    }

#pragma unroll
    for (int mt = 0; mt < 8; ++mt) {          // neuron (M) tiles
      // bias along M: lane needs 8 consecutive values -> two broadcast float4s
      const float4* bp = reinterpret_cast<const float4*>(bias + mt * 16 + (hi << 3));
      float4 bz0 = bp[0], bz1 = bp[1];
      v8f acc0, acc1;
      acc0[0] = bz0.x; acc0[1] = bz0.y; acc0[2] = bz0.z; acc0[3] = bz0.w;
      acc0[4] = bz1.x; acc0[5] = bz1.y; acc0[6] = bz1.z; acc0[7] = bz1.w;
      acc1 = acc0;

#pragma unroll
      for (int kt = 0; kt < 4; ++kt) {
        FragU wf;                              // one weight fragment feeds both N-tiles
        load_frag_lds(wt, mt * 16 + M, WSTR, kt * 32, hi, wf);
        acc0 = __builtin_amdgcn_wmma_f32_16x16x32_f16(
                   false, wf.h, false, bh0[kt].h, (short)0, acc0, false, false);
        acc1 = __builtin_amdgcn_wmma_f32_16x16x32_f16(
                   false, wf.h, false, bh1[kt].h, (short)0, acc1, false, false);
      }

      // tanh + pack + one b128 store per point tile
      uint4 s0, s1;
      s0.x = pk16(fast_tanh(acc0[0]), fast_tanh(acc0[1]));
      s0.y = pk16(fast_tanh(acc0[2]), fast_tanh(acc0[3]));
      s0.z = pk16(fast_tanh(acc0[4]), fast_tanh(acc0[5]));
      s0.w = pk16(fast_tanh(acc0[6]), fast_tanh(acc0[7]));
      s1.x = pk16(fast_tanh(acc1[0]), fast_tanh(acc1[1]));
      s1.y = pk16(fast_tanh(acc1[2]), fast_tanh(acc1[3]));
      s1.z = pk16(fast_tanh(acc1[4]), fast_tanh(acc1[5]));
      s1.w = pk16(fast_tanh(acc1[6]), fast_tanh(acc1[7]));
      int col = mt * 16 + (hi << 3);
      *reinterpret_cast<uint4*>(actw + M * ASTR        + col) = s0;
      *reinterpret_cast<uint4*>(actw + (M + 16) * ASTR + col) = s1;
    }
    __syncthreads();

    if (L < 2) {  // restage next layer's transposed f16 weights: b128 copies
      const uint4* src = (const uint4*)(wts + (L + 1) * HID * HID);
      for (int i = tid; i < HID * HID / 8; i += BLOCK) {
        int row = i >> 4, j = i & 15;
        *(uint4*)(wt + row * WSTR + j * 8) = src[i];
      }
      __syncthreads();
    }
  }

  // ---- layer 5: 128 -> 2; W5/b5 via uniform scalar loads, act via b128 ----
  float sum0 = b5[0], sum1 = b5[1];
#pragma unroll 4
  for (int k0 = 0; k0 < HID; k0 += 8) {
    FragS fs;
    fs.q = *reinterpret_cast<const uint4*>(actw + lane * ASTR + k0);
#pragma unroll
    for (int j = 0; j < 8; ++j) {
      float f = (float)fs.hh[j];
      sum0 = fmaf(f, W5[(k0 + j) * 2 + 0], sum0);
      sum1 = fmaf(f, W5[(k0 + j) * 2 + 1], sum1);
    }
  }

  // ---- postprocess: psi / B mask, u = U*y*psi + B*u_hat, v = B*v_hat ----
  float Uv  = U[0];
  float psi = fast_sigmoid(50.0f * (x - 0.05f)) * fast_sigmoid(50.0f * (0.95f - x));
  float Bq  = x * (1.0f - x) * y * (1.0f - y);
  float u   = fmaf(Bq, sum0, Uv * y * psi);
  float v   = Bq * sum1;
  if (gp < npts) {
    out[gp]        = u;
    out[npts + gp] = v;
  }
}

extern "C" void kernel_launch(void* const* d_in, const int* in_sizes, int n_in,
                              void* d_out, int out_size, void* d_ws, size_t ws_size,
                              hipStream_t stream) {
  const float* xy = (const float*)d_in[0];
  const float* U  = (const float*)d_in[1];
  const float* W1 = (const float*)d_in[2];
  const float* b1 = (const float*)d_in[3];
  const float* W2 = (const float*)d_in[4];
  const float* b2 = (const float*)d_in[5];
  const float* W3 = (const float*)d_in[6];
  const float* b3 = (const float*)d_in[7];
  const float* W4 = (const float*)d_in[8];
  const float* b4 = (const float*)d_in[9];
  const float* W5 = (const float*)d_in[10];
  const float* b5 = (const float*)d_in[11];
  float* out = (float*)d_out;
  _Float16* wts = (_Float16*)d_ws;   // 3 * 128*128 f16 = 96 KB scratch

  int npts = in_sizes[0] / 2;

  // Pass 1: convert + transpose hidden-layer weights to f16 (tiny, L2-resident).
  int nconv = 3 * HID * HID;
  convert_weights_kernel<<<(nconv + 255) / 256, 256, 0, stream>>>(W2, W3, W4, wts);

  // Pass 2: fused MLP + boundary postprocessing.
  int grid = (npts + PTSPB - 1) / PTSPB;
  size_t shmem = (size_t)HID * WSTR * 2            // transposed weight tile (f16)
               + (size_t)NWAVES * MPW * ASTR * 2   // per-wave activations (f16)
               + 384 * sizeof(float);              // b2,b3,b4
  velocity_mlp_kernel<<<grid, BLOCK, shmem, stream>>>(
      xy, U, W1, b1, b2, b3, b4, W5, b5, wts, out, npts);
}
